// SelfAttention_24970939859697
// MI455X (gfx1250) — matrix operand order repaired and checked by reading, not compiled, not run on hardware
//
#include <hip/hip_runtime.h>
#include <hip/hip_bf16.h>
#include <cstdint>

typedef __attribute__((ext_vector_type(16))) _Float16 v16h;
typedef __attribute__((ext_vector_type(8)))  _Float16 v8h;
typedef __attribute__((ext_vector_type(8)))  float    v8f;
typedef __attribute__((ext_vector_type(4)))  uint32_t u32x4;
typedef __attribute__((ext_vector_type(8)))  uint32_t u32x8;

#define WMMA_F32_F16(A, B, C) \
  __builtin_amdgcn_wmma_f32_16x16x32_f16(false, (A), false, (B), (short)0, (C), false, false)

#if __has_builtin(__builtin_amdgcn_s_wait_tensorcnt)
#define WAIT_TENSORCNT(n) __builtin_amdgcn_s_wait_tensorcnt((short)(n))
#else
#define WAIT_TENSORCNT(n) asm volatile("s_wait_tensorcnt %0" ::"i"(n))
#endif

static constexpr int B_ = 16, S_ = 2048, H_ = 128, D_ = 64;

// ---- LDS byte-offset of a __shared__ object (ptrtoint of an AS(3) pointer) ----
typedef __attribute__((address_space(3))) char lds_char;
__device__ inline uint32_t lds_off_u32(void* p) {
  return (uint32_t)(size_t)(lds_char*)p;
}

// ---- TDM: async 2D tile load Global -> LDS (ISA 08, D# per sec 8.3/8.4) ------
// Loads tile_d1 rows x tile_d0 elements (2-byte elements) starting at gptr,
// row stride stride0 (elements), packed contiguously into LDS at lds_off.
__device__ inline void tdm_load_2d_f16(uint32_t lds_off, const _Float16* gptr,
                                       uint32_t tensor_d0, uint32_t tensor_d1,
                                       uint32_t tile_d0, uint32_t tile_d1,
                                       uint32_t stride0) {
  const uint64_t ga = (uint64_t)(uintptr_t)gptr;
  u32x4 g0;
  g0[0] = 1u;                                              // count=1 (valid user D#)
  g0[1] = lds_off;                                         // lds_addr (bytes)
  g0[2] = (uint32_t)ga;                                    // global_addr[31:0]
  g0[3] = (uint32_t)((ga >> 32) & 0x01ffffffu) | (2u << 30); // addr[56:32] | type=2
  u32x8 g1;
  g1[0] = 1u << 16;                                        // data_size=1 (2B), wg_mask=0
  g1[1] = (tensor_d0 & 0xffffu) << 16;                     // tensor_dim0[15:0] @ bits 63:48
  g1[2] = (tensor_d0 >> 16) | ((tensor_d1 & 0xffffu) << 16); // dim0[31:16] | dim1[15:0]
  g1[3] = (tensor_d1 >> 16) | ((tile_d0 & 0xffffu) << 16); // dim1[31:16] | tile_dim0
  g1[4] = tile_d1 & 0xffffu;                               // tile_dim1 (tile_dim2=0)
  g1[5] = stride0;                                         // tensor_dim0_stride[31:0]
  g1[6] = 0u;                                              // stride0[47:32] | dim1_stride lo
  g1[7] = 0u;
  asm volatile("tensor_load_to_lds %0, %1" ::"s"(g0), "s"(g1) : "memory");
}

// ---- fragment loaders (contiguous k = 16*hi + s in A and B, wave32) --------------
__device__ inline v16h load_a_frag(const _Float16* base, int ld, int kbase, int lane) {
  const int lo = lane & 15, hi = lane >> 4;
  const _Float16* p = base + (size_t)lo * ld + kbase + hi * 16;
  v8h x0 = *(const v8h*)(p);
  v8h x1 = *(const v8h*)(p + 8);
  v16h r;
#pragma unroll
  for (int i = 0; i < 8; ++i) { r[i] = x0[i]; r[i + 8] = x1[i]; }
  return r;
}

// B-fragment for Q*K^T: B[d][n] = K[row n, d]; rows of ld halves, contiguous d.
__device__ inline v16h load_b_ktrans(const _Float16* Krows, int ld, int dbase, int lane) {
  const int lo = lane & 15, hi = lane >> 4;
  return *(const v16h*)(Krows + (size_t)lo * ld + dbase + hi * 16);
}

// B-fragment from an f32 weight matrix (row-major [K x N], ld = N).
__device__ inline v16h load_b_w32(const float* W, int ld, int kbase, int n0, int lane) {
  const int lo = lane & 15, hi = lane >> 4;
  const float* p = W + (size_t)(kbase + hi * 16) * ld + n0 + lo;
  v16h r;
#pragma unroll
  for (int j = 0; j < 16; ++j) r[j] = (_Float16)p[(size_t)j * ld];
  return r;
}

// ---- kernel 1: xp = x + PE;  Q,K row-major f16; V transposed f16 -------------
__global__ __launch_bounds__(32) void qkv_pe_kernel(
    const float* __restrict__ x, const float* __restrict__ Wq,
    const float* __restrict__ Wk, const float* __restrict__ Wv,
    _Float16* __restrict__ Qh, _Float16* __restrict__ Kh,
    _Float16* __restrict__ Vt) {
  __shared__ _Float16 xs[16 * 128];
  const int lane = threadIdx.x;
  const int b  = blockIdx.x >> 7;
  const int s0 = (blockIdx.x & 127) << 4;

  for (int idx = lane; idx < 16 * 128; idx += 32) {
    const int r = idx >> 7, c = idx & 127;
    const float freq = __expf(-(float)(c & ~1) * (9.210340371976184f / 128.0f));
    const float ang  = (float)(s0 + r) * freq;
    const float pe   = (c & 1) ? __cosf(ang) : __sinf(ang);
    xs[idx] = (_Float16)(x[((size_t)b * S_ + s0 + r) * H_ + c] + pe);
  }
  __syncthreads();

  const int lo = lane & 15, hi = lane >> 4;
  const float* Wmats[3] = {Wq, Wk, Wv};
#pragma unroll
  for (int m = 0; m < 3; ++m) {
#pragma unroll
    for (int nt = 0; nt < 4; ++nt) {
      v8f acc = {};
#pragma unroll
      for (int kt = 0; kt < 4; ++kt) {
        v16h a  = load_a_frag(xs, 128, kt * 32, lane);
        v16h bb = load_b_w32(Wmats[m], 64, kt * 32, nt * 16, lane);
        acc = WMMA_F32_F16(a, bb, acc);
      }
      const int n = nt * 16 + lo;
#pragma unroll
      for (int r = 0; r < 8; ++r) {
        const int row = s0 + r + 8 * hi;
        const _Float16 h = (_Float16)acc[r];
        if (m == 0)      Qh[((size_t)b * S_ + row) * D_ + n] = h;
        else if (m == 1) Kh[((size_t)b * S_ + row) * D_ + n] = h;
        else             Vt[((size_t)b * D_ + n) * (size_t)S_ + row] = h;
      }
    }
  }
}

// ---- kernel 2: query-axis softmax stats m[b,k], denom[b,k] (online) ----------
__global__ __launch_bounds__(32) void colstats_kernel(
    const _Float16* __restrict__ Qh, const _Float16* __restrict__ Kh,
    float* __restrict__ Ms, float* __restrict__ Ds) {
  const int lane = threadIdx.x;
  const int lo = lane & 15, hi = lane >> 4;
  const int b  = blockIdx.x >> 7;
  const int k0 = (blockIdx.x & 127) << 4;

  const _Float16* Krows = Kh + ((size_t)b * S_ + k0) * D_;
  const v16h bk0 = load_b_ktrans(Krows, 64, 0, lane);
  const v16h bk1 = load_b_ktrans(Krows, 64, 32, lane);

  float m = -3.0e38f, sum = 0.0f;
  for (int qt = 0; qt < 128; ++qt) {
    const _Float16* Qrows = Qh + ((size_t)b * S_ + qt * 16) * D_;
    v16h a0 = load_a_frag(Qrows, 64, 0, lane);
    v16h a1 = load_a_frag(Qrows, 64, 32, lane);
    v8f s = {};
    s = WMMA_F32_F16(a0, bk0, s);
    s = WMMA_F32_F16(a1, bk1, s);

    float sv[8], tmax = -3.0e38f;
#pragma unroll
    for (int r = 0; r < 8; ++r) { sv[r] = s[r] * 0.125f; tmax = fmaxf(tmax, sv[r]); }
    tmax = fmaxf(tmax, __shfl_xor(tmax, 16, 32));
    const float nm = fmaxf(m, tmax);
    float ts = 0.0f;
#pragma unroll
    for (int r = 0; r < 8; ++r) ts += __expf(sv[r] - nm);
    ts += __shfl_xor(ts, 16, 32);
    sum = sum * __expf(m - nm) + ts;
    m = nm;
  }
  if (hi == 0) {
    Ms[(size_t)b * S_ + k0 + lo] = m;
    Ds[(size_t)b * S_ + k0 + lo] = sum;
  }
}

// ---- kernel 3: z = softmax_q(QK^T/8) @ V fused with out = z.Wf + bf ----------
// K and V tiles are TDM double-buffered into LDS; scores and z use WMMA.
__global__ __launch_bounds__(32) void attn_out_kernel(
    const _Float16* __restrict__ Qh, const _Float16* __restrict__ Kh,
    const _Float16* __restrict__ Vt, const float* __restrict__ Ms,
    const float* __restrict__ Ds, const float* __restrict__ Wf,
    const float* __restrict__ bf, float* __restrict__ out) {
  __shared__ _Float16 kt_lds[2][32 * 64];   // 32 k-rows x 64 d
  __shared__ _Float16 vt_lds[2][64 * 32];   // 64 d-rows x 32 k
  __shared__ _Float16 ps[16 * 32];          // P tile transpose staging
  const int lane = threadIdx.x;
  const int lo = lane & 15, hi = lane >> 4;
  const int b  = blockIdx.x >> 7;
  const int q0 = (blockIdx.x & 127) << 4;

  const _Float16* Kbase = Kh + (size_t)b * S_ * D_;
  const _Float16* Vbase = Vt + (size_t)b * D_ * S_;

  const _Float16* Qrows = Qh + ((size_t)b * S_ + q0) * D_;
  const v16h a0 = load_a_frag(Qrows, 64, 0, lane);
  const v16h a1 = load_a_frag(Qrows, 64, 32, lane);

  // prefetch k-block 0 into buffer 0 (TDM, TENSORcnt-tracked)
  tdm_load_2d_f16(lds_off_u32(&kt_lds[0][0]), Kbase, 64, S_, 64, 32, 64);
  tdm_load_2d_f16(lds_off_u32(&vt_lds[0][0]), Vbase, S_, 64, 32, 64, S_);

  v8f z0 = {}, z1 = {}, z2 = {}, z3 = {};
  for (int kk = 0; kk < 64; ++kk) {
    if (kk + 1 < 64) {                       // prefetch next 32-k block
      const int nb = (kk + 1) & 1;
      tdm_load_2d_f16(lds_off_u32(&kt_lds[nb][0]), Kbase + (size_t)(kk + 1) * 32 * 64,
                      64, S_, 64, 32, 64);
      tdm_load_2d_f16(lds_off_u32(&vt_lds[nb][0]), Vbase + (size_t)(kk + 1) * 32,
                      S_, 64, 32, 64, S_);
      WAIT_TENSORCNT(2);                     // current block's 2 DMAs complete
    } else {
      WAIT_TENSORCNT(0);
    }
    const _Float16* kt = &kt_lds[kk & 1][0];
    const _Float16* vt = &vt_lds[kk & 1][0];

#pragma unroll
    for (int sub = 0; sub < 2; ++sub) {
      const int k0 = (kk * 2 + sub) * 16;
      v16h bk0 = load_b_ktrans(kt + (size_t)sub * 16 * 64, 64, 0, lane);
      v16h bk1 = load_b_ktrans(kt + (size_t)sub * 16 * 64, 64, 32, lane);
      v8f s = {};
      s = WMMA_F32_F16(a0, bk0, s);
      s = WMMA_F32_F16(a1, bk1, s);
      const float mk = Ms[(size_t)b * S_ + k0 + lo];
      const float rd = 1.0f / Ds[(size_t)b * S_ + k0 + lo];
#pragma unroll
      for (int r = 0; r < 8; ++r) {          // element (q = r+8*hi, k_local = lo)
        const float p = __expf(s[r] * 0.125f - mk) * rd;
        ps[(r + 8 * hi) * 32 + sub * 16 + lo] = (_Float16)p;
      }
    }
    __syncthreads();
    const v16h pa = load_a_frag(ps, 32, 0, lane);        // P as 16x32 A fragment
    z0 = WMMA_F32_F16(pa, *(const v16h*)(vt + (size_t)(0 * 16 + lo) * 32 + hi * 16), z0);
    z1 = WMMA_F32_F16(pa, *(const v16h*)(vt + (size_t)(1 * 16 + lo) * 32 + hi * 16), z1);
    z2 = WMMA_F32_F16(pa, *(const v16h*)(vt + (size_t)(2 * 16 + lo) * 32 + hi * 16), z2);
    z3 = WMMA_F32_F16(pa, *(const v16h*)(vt + (size_t)(3 * 16 + lo) * 32 + hi * 16), z3);
    __syncthreads();                         // ds reads done before next TDM overwrite
  }

  // out[b, q] = z[q,:] . Wf + bf   (z C-layout: N = nt*16+lo, M = r+8*hi)
  const float wf0 = Wf[0 * 16 + lo], wf1 = Wf[1 * 16 + lo];
  const float wf2 = Wf[2 * 16 + lo], wf3 = Wf[3 * 16 + lo];
  float part[8];
#pragma unroll
  for (int r = 0; r < 8; ++r)
    part[r] = z0[r] * wf0 + z1[r] * wf1 + z2[r] * wf2 + z3[r] * wf3;
#pragma unroll
  for (int r = 0; r < 8; ++r) {
    part[r] += __shfl_xor(part[r], 1, 32);
    part[r] += __shfl_xor(part[r], 2, 32);
    part[r] += __shfl_xor(part[r], 4, 32);
    part[r] += __shfl_xor(part[r], 8, 32);
  }
  if (lo == 0) {
    const float bfv = bf[0];
#pragma unroll
    for (int r = 0; r < 8; ++r)
      out[(size_t)b * S_ + q0 + r + 8 * hi] = part[r] + bfv;
  }
}

extern "C" void kernel_launch(void* const* d_in, const int* in_sizes, int n_in,
                              void* d_out, int out_size, void* d_ws, size_t ws_size,
                              hipStream_t stream) {
  const float* x  = (const float*)d_in[0];
  const float* Wq = (const float*)d_in[1];
  const float* Wk = (const float*)d_in[2];
  const float* Wv = (const float*)d_in[3];
  const float* Wf = (const float*)d_in[4];
  const float* bf = (const float*)d_in[5];
  float* out = (float*)d_out;

  // workspace layout (~12.8 MB): Qh | Kh | Vt (f16) | Ms | Ds (f32)
  const size_t nBS = (size_t)B_ * S_;
  _Float16* Qh = (_Float16*)d_ws;
  _Float16* Kh = Qh + nBS * D_;
  _Float16* Vt = Kh + nBS * D_;
  float*    Ms = (float*)(Vt + nBS * D_);
  float*    Ds = Ms + nBS;

  const int blocks = (int)(nBS / 16);        // 2048 one-wave workgroups
  qkv_pe_kernel <<<blocks, 32, 0, stream>>>(x, Wq, Wk, Wv, Qh, Kh, Vt);
  colstats_kernel<<<blocks, 32, 0, stream>>>(Qh, Kh, Ms, Ds);
  attn_out_kernel<<<blocks, 32, 0, stream>>>(Qh, Kh, Vt, Ms, Ds, Wf, bf, out);
}
